// RelativeMultiHeadAttn_85839216378238
// MI455X (gfx1250) — matrix operand order repaired and checked by reading, not compiled, not run on hardware
//
#include <hip/hip_runtime.h>
#include <hip/hip_bf16.h>

// ---------------------------------------------------------------------------
// Transformer-XL relative multi-head attention for MI455X (gfx1250, wave32).
// All matmuls run on v_wmma_f32_16x16x32_bf16 (f32 accumulate).
// ---------------------------------------------------------------------------

typedef __bf16 bf16_t;
typedef __attribute__((ext_vector_type(16))) __bf16 v16bf;
typedef __attribute__((ext_vector_type(8)))  float  v8f;

#define NHEADS 16
#define DHEAD  64
#define DMODEL 1024
#define QL     512
#define ML     512
#define KL     1024   // QL + ML
#define BSZ    16

static __device__ __forceinline__ v8f wmma_bf16(v16bf a, v16bf b, v8f c) {
  // (neg_a, A, neg_b, B, c_mod, C, reuse_a, reuse_b)
  return __builtin_amdgcn_wmma_f32_16x16x32_bf16(false, a, false, b, (short)0, c, false, false);
}

static __device__ __forceinline__ v8f vzero8() {
  v8f z = {0.f, 0.f, 0.f, 0.f, 0.f, 0.f, 0.f, 0.f};
  return z;
}

// A-operand fragment (16 rows x 32 K), row-major source, leading dim ld.
// A layout: lane l (h=l>>4, r=l&15) holds row r;
// elements e=0..15 -> k0+h*16+e (contiguous, matches frag_b).
static __device__ __forceinline__ v16bf frag_a(const bf16_t* base, int ld,
                                               int row0, int k0, int lane) {
  const int h = lane >> 4, r = lane & 15;
  const bf16_t* p = base + (size_t)(row0 + r) * ld + (k0 + h * 16);
  union { uint4 q[2]; v16bf v; } u;
  u.q[0] = *reinterpret_cast<const uint4*>(p);
  u.q[1] = *reinterpret_cast<const uint4*>(p + 8);
  return u.v;
}

// B-operand fragment for D = A * Y^T, Y row-major (N x K).
// Lane l (h=l>>4, c=l&15) holds Y[col0+c][k0 + h*16 + e], e=0..15 (contiguous).
static __device__ __forceinline__ v16bf frag_b(const bf16_t* base, int ld,
                                               int col0, int k0, int lane) {
  const int h = lane >> 4, c = lane & 15;
  const bf16_t* p = base + (size_t)(col0 + c) * ld + (k0 + h * 16);
  union { uint4 q[2]; v16bf v; } u;
  u.q[0] = *reinterpret_cast<const uint4*>(p);
  u.q[1] = *reinterpret_cast<const uint4*>(p + 8);
  return u.v;
}

// Same as frag_b but clamps the Y row index (used for the shifted R block,
// where out-of-range rows correspond to masked score positions).
static __device__ __forceinline__ v16bf frag_b_clamp(const bf16_t* base, int ld,
                                                     int col0, int k0, int lane,
                                                     int maxrow) {
  const int h = lane >> 4, c = lane & 15;
  int row = col0 + c; if (row > maxrow) row = maxrow; if (row < 0) row = 0;
  const bf16_t* p = base + (size_t)row * ld + (k0 + h * 16);
  union { uint4 q[2]; v16bf v; } u;
  u.q[0] = *reinterpret_cast<const uint4*>(p);
  u.q[1] = *reinterpret_cast<const uint4*>(p + 8);
  return u.v;
}

// ---------------------------------------------------------------------------
// Packing kernels (f32 -> bf16, optional transpose)
// ---------------------------------------------------------------------------
__global__ __launch_bounds__(256) void cvt_kernel(const float* __restrict__ src,
                                                  bf16_t* __restrict__ dst, int n) {
  int i = blockIdx.x * 256 + threadIdx.x;
  if (i < n) dst[i] = (bf16_t)src[i];
}

// dst (cols x rows) = transpose(src (rows x cols)), bf16
__global__ __launch_bounds__(256) void cvt_t_kernel(const float* __restrict__ src,
                                                    bf16_t* __restrict__ dst,
                                                    int rows, int cols) {
  int i = blockIdx.x * 256 + threadIdx.x;
  if (i < rows * cols) {
    int r = i / cols, c = i % cols;
    dst[(size_t)c * rows + r] = (bf16_t)src[i];
  }
}

// ---------------------------------------------------------------------------
// QKV projection: cat(16384 x 1024) @ qkv_w -> Q(+biases)/K/V in WMMA layouts.
// Workgroup = 4 waves (2x2), wave tile 32x64. Grid (256, 24).
// ---------------------------------------------------------------------------
__global__ __launch_bounds__(128) void qkv_gemm_kernel(
    const bf16_t* __restrict__ A,    // cat  [16384][1024]
    const bf16_t* __restrict__ Bt,   // qkvT [3072][1024]
    const float*  __restrict__ rwb,  // r_w_bias [16][64]
    const float*  __restrict__ rrb,  // r_r_bias [16][64]
    bf16_t* __restrict__ RWQ,        // [b][n][512][64]
    bf16_t* __restrict__ RRQ,        // [b][n][512][64]
    bf16_t* __restrict__ Kb,         // [b][n][1024][64]
    bf16_t* __restrict__ VT) {       // [b][n][64][1024]
  const int lane = threadIdx.x & 31;
  const int wave = threadIdx.x >> 5;
  const int row0 = blockIdx.x * 64 + (wave >> 1) * 32;
  const int col0 = blockIdx.y * 128 + (wave & 1) * 64;

  v8f acc[2][4];
  for (int i = 0; i < 2; ++i) for (int j = 0; j < 4; ++j) acc[i][j] = vzero8();

  for (int k0 = 0; k0 < DMODEL; k0 += 32) {
    v16bf a0 = frag_a(A, DMODEL, row0,      k0, lane);
    v16bf a1 = frag_a(A, DMODEL, row0 + 16, k0, lane);
#pragma unroll
    for (int t = 0; t < 4; ++t) {
      v16bf b = frag_b(Bt, DMODEL, col0 + 16 * t, k0, lane);
      acc[0][t] = wmma_bf16(a0, b, acc[0][t]);
      acc[1][t] = wmma_bf16(a1, b, acc[1][t]);
    }
  }

  const int h = lane >> 4, r = lane & 15;
#pragma unroll
  for (int ms = 0; ms < 2; ++ms)
#pragma unroll
    for (int t = 0; t < 4; ++t)
#pragma unroll
      for (int v = 0; v < 8; ++v) {
        int row = row0 + ms * 16 + v + 8 * h;       // (t, b) flattened
        int col = col0 + t * 16 + r;                // 0..3071
        float val = acc[ms][t][v];
        int tt = row >> 4, b = row & 15;
        int sector = col >> 10;
        int hn = (col & 1023) >> 6;
        int d  = col & 63;
        if (sector == 0) {
          if (tt >= ML) {
            int i = tt - ML;
            size_t idx = (((size_t)b * NHEADS + hn) * QL + i) * DHEAD + d;
            RWQ[idx] = (bf16_t)(val + rwb[hn * DHEAD + d]);
            RRQ[idx] = (bf16_t)(val + rrb[hn * DHEAD + d]);
          }
        } else if (sector == 1) {
          Kb[(((size_t)b * NHEADS + hn) * KL + tt) * DHEAD + d] = (bf16_t)val;
        } else {
          VT[(((size_t)b * NHEADS + hn) * DHEAD + d) * KL + tt] = (bf16_t)val;
        }
      }
}

// ---------------------------------------------------------------------------
// r_head_k = r @ r_w, output layout RHK[n][p][64]. Grid (16, 8).
// ---------------------------------------------------------------------------
__global__ __launch_bounds__(128) void rhk_gemm_kernel(
    const bf16_t* __restrict__ A,    // r bf16 [1024][1024]
    const bf16_t* __restrict__ Bt,   // r_wT   [1024][1024]
    bf16_t* __restrict__ RHK) {      // [n][1024][64]
  const int lane = threadIdx.x & 31;
  const int wave = threadIdx.x >> 5;
  const int row0 = blockIdx.x * 64 + (wave >> 1) * 32;
  const int col0 = blockIdx.y * 128 + (wave & 1) * 64;

  v8f acc[2][4];
  for (int i = 0; i < 2; ++i) for (int j = 0; j < 4; ++j) acc[i][j] = vzero8();

  for (int k0 = 0; k0 < DMODEL; k0 += 32) {
    v16bf a0 = frag_a(A, DMODEL, row0,      k0, lane);
    v16bf a1 = frag_a(A, DMODEL, row0 + 16, k0, lane);
#pragma unroll
    for (int t = 0; t < 4; ++t) {
      v16bf b = frag_b(Bt, DMODEL, col0 + 16 * t, k0, lane);
      acc[0][t] = wmma_bf16(a0, b, acc[0][t]);
      acc[1][t] = wmma_bf16(a1, b, acc[1][t]);
    }
  }
  const int h = lane >> 4, r = lane & 15;
#pragma unroll
  for (int ms = 0; ms < 2; ++ms)
#pragma unroll
    for (int t = 0; t < 4; ++t)
#pragma unroll
      for (int v = 0; v < 8; ++v) {
        int p   = row0 + ms * 16 + v + 8 * h;
        int col = col0 + t * 16 + r;
        int hn = col >> 6, d = col & 63;
        RHK[((size_t)hn * KL + p) * DHEAD + d] = (bf16_t)acc[ms][t][v];
      }
}

// ---------------------------------------------------------------------------
// Fused attention (flash-style). Grid (b=16, n=16, itile=8), 4 waves/WG.
// Each wave owns 16 query rows; streams over 64-key chunks.
// rel_shift resolved analytically: BD[i,j] = rr_q[i] . R[j + QL-1-i].
// ---------------------------------------------------------------------------
__global__ __launch_bounds__(128) void attn_kernel(
    const bf16_t* __restrict__ RWQ, const bf16_t* __restrict__ RRQ,
    const bf16_t* __restrict__ Kb,  const bf16_t* __restrict__ VT,
    const bf16_t* __restrict__ RHK, bf16_t* __restrict__ AV) {
  __shared__ __align__(16) float  lds_bd[4][16][80];
  __shared__ __align__(16) bf16_t lds_p[4][16][64];

  const int lane = threadIdx.x & 31;
  const int wave = threadIdx.x >> 5;
  const int h = lane >> 4, r = lane & 15;
  const int bidx = blockIdx.x, nidx = blockIdx.y, itile = blockIdx.z;
  const int i0 = itile * 64 + wave * 16;

  const size_t bn = (size_t)bidx * NHEADS + nidx;
  const bf16_t* Q1 = RWQ + bn * QL * DHEAD;
  const bf16_t* Q2 = RRQ + bn * QL * DHEAD;
  const bf16_t* Kp = Kb  + bn * KL * DHEAD;
  const bf16_t* Vp = VT  + bn * DHEAD * KL;   // [64][1024]
  const bf16_t* Rp = RHK + (size_t)nidx * KL * DHEAD;

  v16bf aRW[2], aRR[2];
  aRW[0] = frag_a(Q1, DHEAD, i0, 0, lane);  aRW[1] = frag_a(Q1, DHEAD, i0, 32, lane);
  aRR[0] = frag_a(Q2, DHEAD, i0, 0, lane);  aRR[1] = frag_a(Q2, DHEAD, i0, 32, lane);

  float m_run[8], l_run[8];
  v8f o[4];
#pragma unroll
  for (int v = 0; v < 8; ++v) { m_run[v] = -3.0e38f; l_run[v] = 0.f; }
#pragma unroll
  for (int dt = 0; dt < 4; ++dt) o[dt] = vzero8();

  // Uniform trip count across the workgroup (required for __syncthreads).
  const int i0max = itile * 64 + 48;
  int jend = ((i0max + 527) / 64 + 1) * 64;
  if (jend > KL) jend = KL;
  const float scale = 0.125f;  // 1/sqrt(64)

  for (int j0 = 0; j0 < jend; j0 += 64) {
    // ---- S = RWQ @ K^T (16 x 64) ----
    v8f s[4];
#pragma unroll
    for (int t = 0; t < 4; ++t) {
      s[t] = vzero8();
      v16bf b0 = frag_b(Kp, DHEAD, j0 + 16 * t, 0, lane);
      s[t] = wmma_bf16(aRW[0], b0, s[t]);
      v16bf b1 = frag_b(Kp, DHEAD, j0 + 16 * t, 32, lane);
      s[t] = wmma_bf16(aRW[1], b1, s[t]);
    }
    // ---- BD wide tile: 16 x 80 over shifted R rows ----
    const int pbase = j0 + (QL - 16) - i0;   // j0 + 496 - i0  (>= 0)
    v8f wv[5];
#pragma unroll
    for (int t = 0; t < 5; ++t) {
      wv[t] = vzero8();
      v16bf b0 = frag_b_clamp(Rp, DHEAD, pbase + 16 * t, 0, lane, KL - 1);
      wv[t] = wmma_bf16(aRR[0], b0, wv[t]);
      v16bf b1 = frag_b_clamp(Rp, DHEAD, pbase + 16 * t, 32, lane, KL - 1);
      wv[t] = wmma_bf16(aRR[1], b1, wv[t]);
    }
#pragma unroll
    for (int t = 0; t < 5; ++t)
#pragma unroll
      for (int v = 0; v < 8; ++v)
        lds_bd[wave][v + 8 * h][16 * t + r] = wv[t][v];
    __syncthreads();

    // ---- combine AC + shifted BD, scale, mask ----
#pragma unroll
    for (int t = 0; t < 4; ++t)
#pragma unroll
      for (int v = 0; v < 8; ++v) {
        int m = v + 8 * h;
        int i = i0 + m;
        int j = j0 + 16 * t + r;
        int c = 16 * t + r + 15 - m;           // (j + QL-1-i) - pbase
        float sc = (s[t][v] + lds_bd[wave][m][c]) * scale;
        if (j > i + ML) sc = -1.0e30f;
        s[t][v] = sc;
      }
    __syncthreads();

    // ---- streaming softmax update ----
#pragma unroll
    for (int v = 0; v < 8; ++v) {
      float mx = fmaxf(fmaxf(s[0][v], s[1][v]), fmaxf(s[2][v], s[3][v]));
#pragma unroll
      for (int off = 1; off < 16; off <<= 1) mx = fmaxf(mx, __shfl_xor(mx, off, 32));
      float m_new = fmaxf(m_run[v], mx);
      float resc = __expf(m_run[v] - m_new);
      float rs = 0.f;
#pragma unroll
      for (int t = 0; t < 4; ++t) {
        float pr = __expf(s[t][v] - m_new);
        s[t][v] = pr;
        rs += pr;
      }
#pragma unroll
      for (int off = 1; off < 16; off <<= 1) rs += __shfl_xor(rs, off, 32);
      l_run[v] = l_run[v] * resc + rs;
      m_run[v] = m_new;
#pragma unroll
      for (int dt = 0; dt < 4; ++dt) o[dt][v] *= resc;
    }

    // ---- P through LDS (bf16) into A-fragment layout, then O += P @ V ----
#pragma unroll
    for (int t = 0; t < 4; ++t)
#pragma unroll
      for (int v = 0; v < 8; ++v)
        lds_p[wave][v + 8 * h][16 * t + r] = (bf16_t)s[t][v];
    __syncthreads();
    v16bf aP0 = frag_a(&lds_p[wave][0][0], 64, 0, 0, lane);
    v16bf aP1 = frag_a(&lds_p[wave][0][0], 64, 0, 32, lane);
#pragma unroll
    for (int dt = 0; dt < 4; ++dt) {
      v16bf b0 = frag_b(Vp, KL, dt * 16, j0, lane);
      o[dt] = wmma_bf16(aP0, b0, o[dt]);
      v16bf b1 = frag_b(Vp, KL, dt * 16, j0 + 32, lane);
      o[dt] = wmma_bf16(aP1, b1, o[dt]);
    }
    __syncthreads();
  }

  // ---- finalize: normalize and write attn_vec (rows = i*BSZ + b) ----
#pragma unroll
  for (int dt = 0; dt < 4; ++dt)
#pragma unroll
    for (int v = 0; v < 8; ++v) {
      int m = v + 8 * h;
      int i = i0 + m;
      int d = dt * 16 + r;
      float val = o[dt][v] / l_run[v];
      AV[((size_t)i * BSZ + bidx) * DMODEL + nidx * DHEAD + d] = (bf16_t)val;
    }
}

// ---------------------------------------------------------------------------
// attn_out = AV @ o_w, plus residual w -> X (f32). Grid (128, 8).
// ---------------------------------------------------------------------------
__global__ __launch_bounds__(128) void out_gemm_kernel(
    const bf16_t* __restrict__ A,    // AV [8192][1024]
    const bf16_t* __restrict__ Bt,   // o_wT [1024][1024]
    const float*  __restrict__ wres, // w [8192][1024]
    float* __restrict__ X) {
  const int lane = threadIdx.x & 31;
  const int wave = threadIdx.x >> 5;
  const int row0 = blockIdx.x * 64 + (wave >> 1) * 32;
  const int col0 = blockIdx.y * 128 + (wave & 1) * 64;

  v8f acc[2][4];
  for (int i = 0; i < 2; ++i) for (int j = 0; j < 4; ++j) acc[i][j] = vzero8();

  for (int k0 = 0; k0 < DMODEL; k0 += 32) {
    v16bf a0 = frag_a(A, DMODEL, row0,      k0, lane);
    v16bf a1 = frag_a(A, DMODEL, row0 + 16, k0, lane);
#pragma unroll
    for (int t = 0; t < 4; ++t) {
      v16bf b = frag_b(Bt, DMODEL, col0 + 16 * t, k0, lane);
      acc[0][t] = wmma_bf16(a0, b, acc[0][t]);
      acc[1][t] = wmma_bf16(a1, b, acc[1][t]);
    }
  }
  const int h = lane >> 4, r = lane & 15;
#pragma unroll
  for (int ms = 0; ms < 2; ++ms)
#pragma unroll
    for (int t = 0; t < 4; ++t)
#pragma unroll
      for (int v = 0; v < 8; ++v) {
        size_t row = row0 + ms * 16 + v + 8 * h;
        size_t col = col0 + t * 16 + r;
        X[row * DMODEL + col] = acc[ms][t][v] + wres[row * DMODEL + col];
      }
}

// ---------------------------------------------------------------------------
// Row-wise LayerNorm over 1024 cols. Grid (8192), 256 threads.
// ---------------------------------------------------------------------------
__global__ __launch_bounds__(256) void ln_kernel(const float* __restrict__ X,
                                                 const float* __restrict__ g,
                                                 const float* __restrict__ b,
                                                 float* __restrict__ out) {
  const int row = blockIdx.x;
  const float* x = X + (size_t)row * DMODEL;
  float v[4];
  float s = 0.f;
#pragma unroll
  for (int q = 0; q < 4; ++q) { v[q] = x[threadIdx.x + q * 256]; s += v[q]; }

  __shared__ float red[8];
  const int lane = threadIdx.x & 31, wv = threadIdx.x >> 5;
#pragma unroll
  for (int off = 16; off >= 1; off >>= 1) s += __shfl_xor(s, off, 32);
  if (lane == 0) red[wv] = s;
  __syncthreads();
  float tot = 0.f;
#pragma unroll
  for (int i = 0; i < 8; ++i) tot += red[i];
  const float mean = tot * (1.f / (float)DMODEL);
  __syncthreads();

  float s2 = 0.f;
#pragma unroll
  for (int q = 0; q < 4; ++q) { float d = v[q] - mean; s2 += d * d; }
#pragma unroll
  for (int off = 16; off >= 1; off >>= 1) s2 += __shfl_xor(s2, off, 32);
  if (lane == 0) red[wv] = s2;
  __syncthreads();
  float tot2 = 0.f;
#pragma unroll
  for (int i = 0; i < 8; ++i) tot2 += red[i];
  const float rstd = rsqrtf(tot2 * (1.f / (float)DMODEL) + 1e-6f);

#pragma unroll
  for (int q = 0; q < 4; ++q) {
    int c = threadIdx.x + q * 256;
    out[(size_t)row * DMODEL + c] = (v[q] - mean) * rstd * g[c] + b[c];
  }
}

// ---------------------------------------------------------------------------
// Host launcher
// ---------------------------------------------------------------------------
extern "C" void kernel_launch(void* const* d_in, const int* in_sizes, int n_in,
                              void* d_out, int out_size, void* d_ws, size_t ws_size,
                              hipStream_t stream) {
  (void)in_sizes; (void)n_in; (void)out_size; (void)ws_size;
  const float* w      = (const float*)d_in[0];   // (512,16,1024)
  const float* r      = (const float*)d_in[1];   // (1024,1024)
  const float* mems   = (const float*)d_in[3];   // (512,16,1024)
  const float* qkv_w  = (const float*)d_in[4];   // (1024,3072)
  const float* r_w    = (const float*)d_in[5];   // (1024,1024)
  const float* o_w    = (const float*)d_in[6];   // (1024,1024)
  const float* r_r_b  = (const float*)d_in[7];   // (16,64)
  const float* r_w_b  = (const float*)d_in[8];   // (16,64)
  const float* ln_g   = (const float*)d_in[9];
  const float* ln_b   = (const float*)d_in[10];
  float* out = (float*)d_out;

  // Workspace carve-up (256B aligned). ~166 MB total; X aliases dead cat.
  char* p = (char*)d_ws;
  auto carve = [&](size_t bytes) { char* q = p; p += (bytes + 255) & ~(size_t)255; return q; };
  bf16_t* cat  = (bf16_t*)carve((size_t)KL * BSZ * DMODEL * 2);       // 33.5 MB
  bf16_t* qkvT = (bf16_t*)carve((size_t)3 * DMODEL * DMODEL * 2);     //  6.3 MB
  bf16_t* rb   = (bf16_t*)carve((size_t)KL * DMODEL * 2);             //  2.1 MB
  bf16_t* rwT  = (bf16_t*)carve((size_t)DMODEL * DMODEL * 2);         //  2.1 MB
  bf16_t* owT  = (bf16_t*)carve((size_t)DMODEL * DMODEL * 2);         //  2.1 MB
  bf16_t* RWQ  = (bf16_t*)carve((size_t)BSZ * NHEADS * QL * DHEAD * 2);   // 16.8 MB
  bf16_t* RRQ  = (bf16_t*)carve((size_t)BSZ * NHEADS * QL * DHEAD * 2);   // 16.8 MB
  bf16_t* Kb   = (bf16_t*)carve((size_t)BSZ * NHEADS * KL * DHEAD * 2);   // 33.5 MB
  bf16_t* VT   = (bf16_t*)carve((size_t)BSZ * NHEADS * DHEAD * KL * 2);   // 33.5 MB
  bf16_t* RHK  = (bf16_t*)carve((size_t)NHEADS * KL * DHEAD * 2);         //  2.1 MB
  bf16_t* AV   = (bf16_t*)carve((size_t)QL * BSZ * DMODEL * 2);           // 16.8 MB
  float*  X    = (float*)cat;  // residual buffer reuses cat after QKV GEMM

  const int nwm = ML * BSZ * DMODEL;   // 8,388,608
  // 1) pack inputs to bf16 (cat = [mems ; w] row-major over (t,b))
  cvt_kernel<<<nwm / 256, 256, 0, stream>>>(mems, cat, nwm);
  cvt_kernel<<<nwm / 256, 256, 0, stream>>>(w, cat + nwm, nwm);
  cvt_kernel<<<(KL * DMODEL) / 256, 256, 0, stream>>>(r, rb, KL * DMODEL);
  cvt_t_kernel<<<(DMODEL * 3 * DMODEL) / 256, 256, 0, stream>>>(qkv_w, qkvT, DMODEL, 3 * DMODEL);
  cvt_t_kernel<<<(DMODEL * DMODEL) / 256, 256, 0, stream>>>(r_w, rwT, DMODEL, DMODEL);
  cvt_t_kernel<<<(DMODEL * DMODEL) / 256, 256, 0, stream>>>(o_w, owT, DMODEL, DMODEL);

  // 2) QKV projection + head packing (note: RWQ uses r_w_bias, RRQ uses r_r_bias)
  qkv_gemm_kernel<<<dim3(256, 24), 128, 0, stream>>>(cat, qkvT, r_w_b, r_r_b,
                                                     RWQ, RRQ, Kb, VT);
  // 3) positional projection
  rhk_gemm_kernel<<<dim3(16, 8), 128, 0, stream>>>(rb, rwT, RHK);

  // 4) fused rel-attention
  attn_kernel<<<dim3(BSZ, NHEADS, QL / 64), 128, 0, stream>>>(RWQ, RRQ, Kb, VT, RHK, AV);

  // 5) output projection + residual
  out_gemm_kernel<<<dim3(128, 8), 128, 0, stream>>>(AV, owT, w, X);

  // 6) layernorm
  ln_kernel<<<QL * BSZ, 256, 0, stream>>>(X, ln_g, ln_b, out);
}
